// MambaLayer2_Click_67001489817883
// MI455X (gfx1250) — matrix operand order repaired and checked by reading, not compiled, not run on hardware
//
#include <hip/hip_runtime.h>

// ---------------------------------------------------------------------------
// MambaLayer2_Click forward for gfx1250 (MI455X, wave32, WMMA bf16)
//
// Pipeline: pemb -> LN+packA -> packW1 -> GEMM1 (WMMA, 2x2 blocked)
//        -> conv+SiLU -> chunked-SSD: S1 (per-chunk WMMA: G, M@X, S_c)
//        -> S2 (tiny sequential chunk-state combine)
//        -> S3 (C@H WMMA + gate) -> RMS+packA -> packW2
//        -> GEMM2 (WMMA, 2x2 blocked, fused output transpose)
// ---------------------------------------------------------------------------

typedef __attribute__((ext_vector_type(16))) __bf16 v16bf;
typedef __attribute__((ext_vector_type(8)))  float  v8f;

namespace {
constexpr int   kB      = 4;
constexpr int   kL      = 4104;
constexpr int   kNTOK   = 4096;
constexpr int   kM      = kB * kL;          // 16416 (=1026*16)
constexpr int   kMT     = kM / 16;          // 1026
constexpr int   kDIM    = 320;
constexpr int   kDIN    = 640;
constexpr int   kCONV   = 704;
constexpr int   kDTOFF  = 1344;             // 2*640 + 64
constexpr int   kNPROJ  = 1352;
constexpr int   kNPAD   = 1376;             // 86*16 (even tile count for 2x2)
constexpr int   kNT1    = kNPAD / 16;       // 86
constexpr int   kKT1    = kDIM / 32;        // 10
constexpr int   kNT2    = kDIM / 16;        // 20
constexpr int   kKT2    = kDIN / 32;        // 20
constexpr int   kNH     = 8;
constexpr int   kHD     = 80;
constexpr int   kS      = 128;              // SSD chunk length
constexpr int   kNC     = 33;               // chunks per sequence (last = 8 rows)

// ---- workspace layout (all offsets 256-aligned) ----
constexpr size_t SZ_ZX    = (size_t)kM * kNPAD * 4;       // 90,353,664
constexpr size_t SZ_XBC   = (size_t)kM * kCONV * 4;       // 46,227,456 (also Y_intra)
constexpr size_t SZ_SB    = (size_t)32 * kNC * 2560 * 4;  // 10,813,440
constexpr size_t SZ_CS    = (size_t)32 * kNC * kS * 4;    // 540,672
constexpr size_t OFF_SB   = 0;
constexpr size_t OFF_HB   = OFF_SB + SZ_SB;
constexpr size_t OFF_EB   = OFF_HB + SZ_SB;
constexpr size_t OFF_CS   = OFF_EB + 4352;                // Ebuf: 32*33*4 rounded
constexpr size_t SZ_SSD   = OFF_CS + SZ_CS;
constexpr size_t SZ_PACKA = (size_t)kM * kDIN * 2;        // 21,012,480
constexpr size_t SZ_SCR   = SZ_SSD > SZ_PACKA ? SZ_SSD : SZ_PACKA;
constexpr size_t WS_ZX    = 0;
constexpr size_t WS_XBC   = WS_ZX + SZ_ZX;
constexpr size_t WS_SCR   = WS_XBC + SZ_XBC;  // packA (GEMM phases) | SSD scratch
constexpr size_t WS_PACKW = WS_SCR + SZ_SCR;
constexpr size_t WS_PEMB  = WS_PACKW + (size_t)kNPAD * kDIM * 2;
} // namespace

union Frag16 { v16bf v; uint4 q[2]; };

__device__ __forceinline__ unsigned short f32_to_bf16(float f) {
  union { float f; unsigned int u; } v; v.f = f;
  unsigned int u = v.u;
  u += 0x7FFFu + ((u >> 16) & 1u);          // round-to-nearest-even
  return (unsigned short)(u >> 16);
}
__device__ __forceinline__ float bf16_to_f32(unsigned short u) {
  union { unsigned int i; float f; } v; v.i = ((unsigned int)u) << 16; return v.f;
}
__device__ __forceinline__ float silu_f(float x) {
  return x / (1.f + __expf(-x));
}

// ---------------------------------------------------------------------------
// K0: point embeddings (32 tokens x 320)
// ---------------------------------------------------------------------------
__global__ void pemb_kernel(const float* __restrict__ coords,
                            const int*   __restrict__ labels,
                            const float* __restrict__ pe_gauss,
                            const float* __restrict__ point_embed,
                            const float* __restrict__ not_a_point,
                            float* __restrict__ pemb)
{
  int t = blockIdx.x, tid = threadIdx.x;
  float c0 = coords[t * 3 + 0] * (2.f / 16.f) - 1.f;
  float c1 = coords[t * 3 + 1] * (2.f / 16.f) - 1.f;
  float c2 = coords[t * 3 + 2] * (2.f / 16.f) - 1.f;
  int j = (tid < 160) ? tid : tid - 160;
  float pe = 6.283185307179586f *
             (c0 * pe_gauss[j] + c1 * pe_gauss[160 + j] + c2 * pe_gauss[320 + j]);
  float val = (tid < 160) ? __sinf(pe) : __cosf(pe);
  int lbl = labels[t];
  float outv;
  if (lbl == -1) outv = not_a_point[tid];
  else {
    int safe = lbl < 0 ? 0 : (lbl > 1 ? 1 : lbl);
    outv = val + point_embed[safe * kDIM + tid];
  }
  pemb[t * kDIM + tid] = outv;
}

// ---------------------------------------------------------------------------
// K1: LayerNorm(320) + pack bf16 WMMA-A frags. One wave per token.
// A-frag (16x32 bf16): lane = m%16 + 16*((k>>3)&1), elem = (k&7)+8*(k>>4)
// ---------------------------------------------------------------------------
__global__ void ln_pack_kernel(const float* __restrict__ x,
                               const float* __restrict__ pemb,
                               const float* __restrict__ ln_w,
                               const float* __restrict__ ln_b,
                               unsigned short* __restrict__ packA)
{
  int wave = threadIdx.x >> 5, lane = threadIdx.x & 31;
  int m = blockIdx.x * 8 + wave;
  if (m >= kM) return;
  int b = m / kL, l = m % kL;
  float v[10]; float s = 0.f, s2 = 0.f;
#pragma unroll
  for (int i = 0; i < 10; ++i) {
    int k = lane + 32 * i;
    float val = (l < kNTOK) ? x[((long)b * kDIM + k) * kNTOK + l]
                            : pemb[(b * 8 + (l - kNTOK)) * kDIM + k];
    v[i] = val; s += val; s2 += val * val;
  }
#pragma unroll
  for (int off = 16; off; off >>= 1) {
    s  += __shfl_xor(s,  off, 32);
    s2 += __shfl_xor(s2, off, 32);
  }
  float mu  = s * (1.f / kDIM);
  float var = s2 * (1.f / kDIM) - mu * mu;
  float inv = rsqrtf(var + 1e-5f);
  int mt = m >> 4, ml = m & 15;
  int fl = ml + 16 * ((lane >> 3) & 1);
  int fe = (lane & 7) + 8 * (lane >> 4);
#pragma unroll
  for (int i = 0; i < 10; ++i) {
    int k = lane + 32 * i;                  // kt = i, klocal = lane
    float xn = (v[i] - mu) * inv * ln_w[k] + ln_b[k];
    packA[(((long)(mt * kKT1 + i) * 32 + fl) << 4) + fe] = f32_to_bf16(xn);
  }
}

// ---------------------------------------------------------------------------
// Weight pack kernels. B-frag (32x16): lane = n%16 + 16*((k>>3)&1), elem = (k&7)+8*(k>>4)
// ---------------------------------------------------------------------------
__global__ void pack_w1_kernel(const float* __restrict__ w,
                               unsigned short* __restrict__ packW)
{
  long idx = (long)blockIdx.x * 256 + threadIdx.x;
  if (idx >= (long)kNPAD * kDIM) return;
  int k = (int)(idx % kDIM);
  int n = (int)(idx / kDIM);
  float val = (n < kNPROJ) ? w[(long)n * kDIM + k] : 0.f;
  int nt = n >> 4, nl = n & 15, kt = k >> 5, kl = k & 31;
  int fl = nl + 16 * ((kl >> 3) & 1), fe = (kl & 7) + 8 * (kl >> 4);
  packW[(((long)(nt * kKT1 + kt) * 32 + fl) << 4) + fe] = f32_to_bf16(val);
}

__global__ void pack_w2_kernel(const float* __restrict__ w,
                               unsigned short* __restrict__ packW)
{
  long idx = (long)blockIdx.x * 256 + threadIdx.x;
  if (idx >= (long)kDIM * kDIN) return;
  int k = (int)(idx % kDIN);
  int n = (int)(idx / kDIN);
  float val = w[(long)n * kDIN + k];
  int nt = n >> 4, nl = n & 15, kt = k >> 5, kl = k & 31;
  int fl = nl + 16 * ((kl >> 3) & 1), fe = (kl & 7) + 8 * (kl >> 4);
  packW[(((long)(nt * kKT2 + kt) * 32 + fl) << 4) + fe] = f32_to_bf16(val);
}

// ---------------------------------------------------------------------------
// GEMM1: zx[16416,1376] = A[16416,320] x W[320,1376]; 32x32 per wave (2x2).
// ---------------------------------------------------------------------------
__device__ __forceinline__ void store_tile_zx(float* zx, int mt, int nt,
                                              int lane, const v8f& acc) {
  int mBase = mt * 16 + (lane >> 4) * 8;
  int n = nt * 16 + (lane & 15);
  float* out = zx + (long)mBase * kNPAD + n;
#pragma unroll
  for (int r = 0; r < 8; ++r) out[(long)r * kNPAD] = acc[r];
}

__global__ void gemm1_kernel(const unsigned short* __restrict__ packA,
                             const unsigned short* __restrict__ packW,
                             float* __restrict__ zx)
{
  int wave = threadIdx.x >> 5, lane = threadIdx.x & 31;
  long t2 = (long)blockIdx.x * 8 + wave;
  if (t2 >= (long)(kMT / 2) * (kNT1 / 2)) return;
  int mt0 = (int)(t2 / (kNT1 / 2)) * 2;
  int nt0 = (int)(t2 % (kNT1 / 2)) * 2;
  v8f a00 = {}, a01 = {}, a10 = {}, a11 = {};
  const v16bf* pa0 = reinterpret_cast<const v16bf*>(packA) + (long)(mt0    ) * kKT1 * 32 + lane;
  const v16bf* pa1 = reinterpret_cast<const v16bf*>(packA) + (long)(mt0 + 1) * kKT1 * 32 + lane;
  const v16bf* pb0 = reinterpret_cast<const v16bf*>(packW) + (long)(nt0    ) * kKT1 * 32 + lane;
  const v16bf* pb1 = reinterpret_cast<const v16bf*>(packW) + (long)(nt0 + 1) * kKT1 * 32 + lane;
#pragma unroll
  for (int kt = 0; kt < kKT1; ++kt) {
    v16bf va0 = pa0[kt * 32], va1 = pa1[kt * 32];
    v16bf vb0 = pb0[kt * 32], vb1 = pb1[kt * 32];
    a00 = __builtin_amdgcn_wmma_f32_16x16x32_bf16(false, va0, false, vb0, (short)0, a00, false, false);
    a01 = __builtin_amdgcn_wmma_f32_16x16x32_bf16(false, va0, false, vb1, (short)0, a01, false, false);
    a10 = __builtin_amdgcn_wmma_f32_16x16x32_bf16(false, va1, false, vb0, (short)0, a10, false, false);
    a11 = __builtin_amdgcn_wmma_f32_16x16x32_bf16(false, va1, false, vb1, (short)0, a11, false, false);
  }
  store_tile_zx(zx, mt0,     nt0,     lane, a00);
  store_tile_zx(zx, mt0,     nt0 + 1, lane, a01);
  store_tile_zx(zx, mt0 + 1, nt0,     lane, a10);
  store_tile_zx(zx, mt0 + 1, nt0 + 1, lane, a11);
}

// ---------------------------------------------------------------------------
// conv: depthwise causal conv-4 + bias + SiLU over xBC cols (640..1343)
// ---------------------------------------------------------------------------
__global__ void conv_silu_kernel(const float* __restrict__ zx,
                                 const float* __restrict__ conv_w,
                                 const float* __restrict__ conv_b,
                                 float* __restrict__ xbc)
{
  long idx = (long)blockIdx.x * 256 + threadIdx.x;
  if (idx >= (long)kM * kCONV) return;
  int  c = (int)(idx % kCONV);
  long m = idx / kCONV;
  int  l = (int)(m % kL);
  float acc = conv_b[c];
#pragma unroll
  for (int j = 0; j < 4; ++j) {
    int dl = l - 3 + j;
    if (dl >= 0)
      acc += zx[(m - 3 + j) * kNPAD + kDIN + c] * conv_w[c * 4 + j];
  }
  xbc[m * kCONV + c] = silu_f(acc);
}

// ---------------------------------------------------------------------------
// S1: per-(b,head,chunk) SSD kernel. All matrix work on v_wmma.
//   G = C·Bᵀ (128x128,K=32); M[i,j]=exp(cs_i-cs_j)·dt_j·G (i>=j)
//   Y_intra = M·X (128x80,K=128) + Dp·x  -> stored over x-cols of xbc
//   S_c = X·(w∘B) (80x32,K=128); E_c = exp(cs_127)
// Fragments built from LDS with contiguous 16B ds_load_b128 reads.
// ---------------------------------------------------------------------------
__global__ __launch_bounds__(256) void ssd_chunk_kernel(
    const float* __restrict__ zx, float* xbc,
    const float* __restrict__ dt_bias, const float* __restrict__ A_log,
    const float* __restrict__ Dp,
    float* __restrict__ Sbuf, float* __restrict__ Ebuf,
    float* __restrict__ csumBuf)
{
  __shared__ float sDt[kS];
  __shared__ float sCs[kS];
  __shared__ float sW[kS];
  __shared__ unsigned short __attribute__((aligned(16))) sXT[80][kS];   // X^T
  __shared__ unsigned short __attribute__((aligned(16))) sB[kS][32];
  __shared__ unsigned short __attribute__((aligned(16))) sBwT[32][kS];  // (w∘B)^T
  __shared__ unsigned short __attribute__((aligned(16))) sC[kS][32];
  __shared__ unsigned short __attribute__((aligned(16))) sM[kS][kS];

  int c  = blockIdx.x % kNC;
  int bh = blockIdx.x / kNC;
  int b  = bh >> 3, hh = bh & 7;
  int tid = threadIdx.x, lane = tid & 31, wave = tid >> 5;
  int valid = kL - c * kS; if (valid > kS) valid = kS;
  long mbase = (long)b * kL + c * kS;
  float Ah  = -__expf(A_log[hh]);
  float dtb = dt_bias[hh];
  float Dph = Dp[hh];

  // ---- stage A: global -> LDS (f32 -> bf16), zero-padded past `valid` ----
  if (tid < kS) {
    float dt = 0.f;
    if (tid < valid) {
      float raw = zx[(mbase + tid) * kNPAD + kDTOFF + hh] + dtb;
      dt = raw > 20.f ? raw : log1pf(__expf(raw));
    }
    sDt[tid] = dt;
  }
  for (int idx = tid; idx < 80 * kS; idx += 256) {
    int p = idx / kS, i = idx % kS;
    float v = (i < valid) ? xbc[(mbase + i) * kCONV + hh * kHD + p] : 0.f;
    sXT[p][i] = f32_to_bf16(v);
  }
  for (int idx = tid; idx < kS * 32; idx += 256) {
    int i = idx / 32, n = idx % 32;
    float vb = (i < valid) ? xbc[(mbase + i) * kCONV + kDIN + n] : 0.f;
    float vc = (i < valid) ? xbc[(mbase + i) * kCONV + kDIN + 32 + n] : 0.f;
    sB[i][n] = f32_to_bf16(vb);
    sC[i][n] = f32_to_bf16(vc);
  }
  __syncthreads();

  // ---- inclusive cumsum of log-decay (serial; 128 adds, negligible) ----
  if (tid == 0) {
    float cs = 0.f;
    for (int i = 0; i < kS; ++i) { cs += sDt[i] * Ah; sCs[i] = cs; }
  }
  __syncthreads();
  float csLast = sCs[kS - 1];
  if (tid < kS) {
    sW[tid] = __expf(csLast - sCs[tid]) * sDt[tid];
    csumBuf[((long)bh * kNC + c) * kS + tid] = sCs[tid];
  }
  if (tid == 0) Ebuf[bh * kNC + c] = __expf(csLast);
  __syncthreads();
  for (int idx = tid; idx < 32 * kS; idx += 256) {
    int n = idx / kS, j = idx % kS;
    sBwT[n][j] = f32_to_bf16(sW[j] * bf16_to_f32(sB[j][n]));
  }
  __syncthreads();

  int h16   = (lane >> 4) ? 16 : 0;
  int h32   = (lane >> 4) ? 32 : 0;
  int mrow  = lane & 15;
  int rbase = (lane >> 4) * 8;

  // ---- G = C·Bᵀ, mask + decay-scale -> M (bf16 in LDS) ----
  for (int t = wave; t < 64; t += 8) {
    int it = t >> 3, jt = t & 7;
    Frag16 fa, fb;
    const char* rowA = (const char*)&sC[it * 16 + mrow][0];
    fa.q[0] = *(const uint4*)(rowA + h16);
    fa.q[1] = *(const uint4*)(rowA + 32 + h16);
    const char* rowB = (const char*)&sB[jt * 16 + mrow][0];
    fb.q[0] = *(const uint4*)(rowB + h16);
    fb.q[1] = *(const uint4*)(rowB + 32 + h16);
    v8f acc = {};
    acc = __builtin_amdgcn_wmma_f32_16x16x32_bf16(false, fa.v, false, fb.v, (short)0, acc, false, false);
#pragma unroll
    for (int r = 0; r < 8; ++r) {
      int ig = it * 16 + rbase + r;
      int jg = jt * 16 + mrow;
      float val = 0.f;
      if (ig >= jg) val = __expf(sCs[ig] - sCs[jg]) * sDt[jg] * acc[r];
      sM[ig][jg] = f32_to_bf16(val);
    }
  }
  __syncthreads();

  // ---- Y_intra = M·X + Dp·x  (128x80, K=128) -> xbc x-cols (in place) ----
  for (int t = wave; t < 40; t += 8) {
    int it = t / 5, pt = t % 5;
    v8f acc = {};
#pragma unroll
    for (int kc = 0; kc < 4; ++kc) {
      Frag16 fa, fb;
      const char* rowA = (const char*)&sM[it * 16 + mrow][kc * 32];
      fa.q[0] = *(const uint4*)(rowA + h16);
      fa.q[1] = *(const uint4*)(rowA + 32 + h16);
      const char* rowB = (const char*)&sXT[pt * 16 + mrow][kc * 32];
      fb.q[0] = *(const uint4*)(rowB + h16);
      fb.q[1] = *(const uint4*)(rowB + 32 + h16);
      acc = __builtin_amdgcn_wmma_f32_16x16x32_bf16(false, fa.v, false, fb.v, (short)0, acc, false, false);
    }
#pragma unroll
    for (int r = 0; r < 8; ++r) {
      int ig = it * 16 + rbase + r;
      int pg = pt * 16 + mrow;
      if (ig < valid) {
        float xv = bf16_to_f32(sXT[pg][ig]);
        xbc[(mbase + ig) * kCONV + hh * kHD + pg] = acc[r] + Dph * xv;
      }
    }
  }

  // ---- S_c = X·(w∘B)  (80x32, K=128) ----
  for (int t = wave; t < 10; t += 8) {
    int pt = t >> 1, nt = t & 1;
    v8f acc = {};
#pragma unroll
    for (int kc = 0; kc < 4; ++kc) {
      Frag16 fa, fb;
      const char* rowA = (const char*)&sXT[pt * 16 + mrow][kc * 32];
      fa.q[0] = *(const uint4*)(rowA + h16);
      fa.q[1] = *(const uint4*)(rowA + 32 + h16);
      const char* rowB = (const char*)&sBwT[nt * 16 + mrow][kc * 32];
      fb.q[0] = *(const uint4*)(rowB + h16);
      fb.q[1] = *(const uint4*)(rowB + 32 + h16);
      acc = __builtin_amdgcn_wmma_f32_16x16x32_bf16(false, fa.v, false, fb.v, (short)0, acc, false, false);
    }
    long sb = ((long)bh * kNC + c) * 2560;
#pragma unroll
    for (int r = 0; r < 8; ++r) {
      int pg = pt * 16 + rbase + r;
      int ng = nt * 16 + mrow;
      Sbuf[sb + pg * 32 + ng] = acc[r];
    }
  }
}

// ---------------------------------------------------------------------------
// S2: sequential chunk-state combine (33 steps, elementwise). 32 blocks.
//   Hbuf[c] = h (state entering chunk c); h = E_c*h + S_c
// ---------------------------------------------------------------------------
__global__ void ssd_state_kernel(const float* __restrict__ Sbuf,
                                 const float* __restrict__ Ebuf,
                                 float* __restrict__ Hbuf)
{
  int bh = blockIdx.x, tid = threadIdx.x;
  float h[10];
#pragma unroll
  for (int i = 0; i < 10; ++i) h[i] = 0.f;
  for (int c = 0; c < kNC; ++c) {
    float E = Ebuf[bh * kNC + c];
    long base = ((long)bh * kNC + c) * 2560;
#pragma unroll
    for (int i = 0; i < 10; ++i) {
      long e = base + tid + 256 * i;
      Hbuf[e] = h[i];
      h[i] = E * h[i] + Sbuf[e];
    }
  }
}

// ---------------------------------------------------------------------------
// S3: Y += exp(cs_i)·(C·H_prevᵀ)  (128x80, K=32 WMMA), then gate with silu(z)
//     and write gated result over the z-columns of zx.
// ---------------------------------------------------------------------------
__global__ __launch_bounds__(256) void ssd_final_kernel(
    float* zx, const float* xbc,
    const float* __restrict__ Hbuf, const float* __restrict__ csumBuf)
{
  __shared__ unsigned short __attribute__((aligned(16))) sC[kS][32];
  __shared__ unsigned short __attribute__((aligned(16))) sH[80][32];
  __shared__ float sCs[kS];

  int c  = blockIdx.x % kNC;
  int bh = blockIdx.x / kNC;
  int b  = bh >> 3, hh = bh & 7;
  int tid = threadIdx.x, lane = tid & 31, wave = tid >> 5;
  int valid = kL - c * kS; if (valid > kS) valid = kS;
  long mbase = (long)b * kL + c * kS;

  for (int idx = tid; idx < kS * 32; idx += 256) {
    int i = idx / 32, n = idx % 32;
    float vc = (i < valid) ? xbc[(mbase + i) * kCONV + kDIN + 32 + n] : 0.f;
    sC[i][n] = f32_to_bf16(vc);
  }
  long hb = ((long)bh * kNC + c) * 2560;
  for (int idx = tid; idx < 80 * 32; idx += 256)
    sH[idx / 32][idx % 32] = f32_to_bf16(Hbuf[hb + idx]);
  if (tid < kS) sCs[tid] = csumBuf[((long)bh * kNC + c) * kS + tid];
  __syncthreads();

  int h16 = (lane >> 4) ? 16 : 0;
  int h32 = (lane >> 4) ? 32 : 0;
  int mrow = lane & 15, rbase = (lane >> 4) * 8;
  for (int t = wave; t < 40; t += 8) {
    int it = t / 5, pt = t % 5;
    Frag16 fa, fb;
    const char* rowA = (const char*)&sC[it * 16 + mrow][0];
    fa.q[0] = *(const uint4*)(rowA + h16);
    fa.q[1] = *(const uint4*)(rowA + 32 + h16);
    const char* rowB = (const char*)&sH[pt * 16 + mrow][0];
    fb.q[0] = *(const uint4*)(rowB + h16);
    fb.q[1] = *(const uint4*)(rowB + 32 + h16);
    v8f acc = {};
    acc = __builtin_amdgcn_wmma_f32_16x16x32_bf16(false, fa.v, false, fb.v, (short)0, acc, false, false);
#pragma unroll
    for (int r = 0; r < 8; ++r) {
      int ig = it * 16 + rbase + r;
      int pg = pt * 16 + mrow;
      if (ig < valid) {
        long m = mbase + ig;
        float y = xbc[m * kCONV + hh * kHD + pg] + __expf(sCs[ig]) * acc[r];
        long zo = m * kNPAD + hh * kHD + pg;
        float z = zx[zo];
        zx[zo] = y * silu_f(z);
      }
    }
  }
}

// ---------------------------------------------------------------------------
// RMSNorm(640) + pack bf16 A-frags for GEMM2. One wave per token.
// ---------------------------------------------------------------------------
__global__ void rms_pack_kernel(const float* __restrict__ zx,
                                const float* __restrict__ rms_w,
                                unsigned short* __restrict__ packA)
{
  int wave = threadIdx.x >> 5, lane = threadIdx.x & 31;
  int m = blockIdx.x * 8 + wave;
  if (m >= kM) return;
  float v[20]; float s2 = 0.f;
#pragma unroll
  for (int i = 0; i < 20; ++i) {
    int k = lane + 32 * i;
    float val = zx[(long)m * kNPAD + k];
    v[i] = val; s2 += val * val;
  }
#pragma unroll
  for (int off = 16; off; off >>= 1) s2 += __shfl_xor(s2, off, 32);
  float inv = rsqrtf(s2 * (1.f / kDIN) + 1e-5f);
  int mt = m >> 4, ml = m & 15;
  int fl = ml + 16 * ((lane >> 3) & 1);
  int fe = (lane & 7) + 8 * (lane >> 4);
#pragma unroll
  for (int i = 0; i < 20; ++i) {
    int k = lane + 32 * i;
    float a = v[i] * inv * rms_w[k];
    packA[(((long)(mt * kKT2 + i) * 32 + fl) << 4) + fe] = f32_to_bf16(a);
  }
}

// ---------------------------------------------------------------------------
// GEMM2: out = A2[16416,640] x W2[640,320]; 2x2 blocked; epilogue fuses the
// (B,L,C)->(B,C,16^3) transpose and drops the 8 point tokens per batch.
// ---------------------------------------------------------------------------
__device__ __forceinline__ void store_tile_out(float* out, int mt, int nt,
                                               int lane, const v8f& acc) {
  int mBase = mt * 16 + (lane >> 4) * 8;
  int n = nt * 16 + (lane & 15);
#pragma unroll
  for (int r = 0; r < 8; ++r) {
    int m = mBase + r;
    int b = m / kL, l = m % kL;
    if (l < kNTOK)
      out[((long)b * kDIM + n) * kNTOK + l] = acc[r];
  }
}

__global__ void gemm2_kernel(const unsigned short* __restrict__ packA,
                             const unsigned short* __restrict__ packW,
                             float* __restrict__ out)
{
  int wave = threadIdx.x >> 5, lane = threadIdx.x & 31;
  long t2 = (long)blockIdx.x * 8 + wave;
  if (t2 >= (long)(kMT / 2) * (kNT2 / 2)) return;
  int mt0 = (int)(t2 / (kNT2 / 2)) * 2;
  int nt0 = (int)(t2 % (kNT2 / 2)) * 2;
  v8f a00 = {}, a01 = {}, a10 = {}, a11 = {};
  const v16bf* pa0 = reinterpret_cast<const v16bf*>(packA) + (long)(mt0    ) * kKT2 * 32 + lane;
  const v16bf* pa1 = reinterpret_cast<const v16bf*>(packA) + (long)(mt0 + 1) * kKT2 * 32 + lane;
  const v16bf* pb0 = reinterpret_cast<const v16bf*>(packW) + (long)(nt0    ) * kKT2 * 32 + lane;
  const v16bf* pb1 = reinterpret_cast<const v16bf*>(packW) + (long)(nt0 + 1) * kKT2 * 32 + lane;
#pragma unroll
  for (int kt = 0; kt < kKT2; ++kt) {
    v16bf va0 = pa0[kt * 32], va1 = pa1[kt * 32];
    v16bf vb0 = pb0[kt * 32], vb1 = pb1[kt * 32];
    a00 = __builtin_amdgcn_wmma_f32_16x16x32_bf16(false, va0, false, vb0, (short)0, a00, false, false);
    a01 = __builtin_amdgcn_wmma_f32_16x16x32_bf16(false, va0, false, vb1, (short)0, a01, false, false);
    a10 = __builtin_amdgcn_wmma_f32_16x16x32_bf16(false, va1, false, vb0, (short)0, a10, false, false);
    a11 = __builtin_amdgcn_wmma_f32_16x16x32_bf16(false, va1, false, vb1, (short)0, a11, false, false);
  }
  store_tile_out(out, mt0,     nt0,     lane, a00);
  store_tile_out(out, mt0,     nt0 + 1, lane, a01);
  store_tile_out(out, mt0 + 1, nt0,     lane, a10);
  store_tile_out(out, mt0 + 1, nt0 + 1, lane, a11);
}

// ---------------------------------------------------------------------------
extern "C" void kernel_launch(void* const* d_in, const int* in_sizes, int n_in,
                              void* d_out, int out_size, void* d_ws, size_t ws_size,
                              hipStream_t stream) {
  (void)in_sizes; (void)n_in; (void)out_size; (void)ws_size;
  const float* x           = (const float*)d_in[0];
  const float* coords      = (const float*)d_in[1];
  const float* ln_w        = (const float*)d_in[2];
  const float* ln_b        = (const float*)d_in[3];
  const float* in_proj_w   = (const float*)d_in[4];
  const float* conv_w      = (const float*)d_in[5];
  const float* conv_b      = (const float*)d_in[6];
  const float* dt_bias     = (const float*)d_in[7];
  const float* A_log       = (const float*)d_in[8];
  const float* Dp          = (const float*)d_in[9];
  const float* rms_w       = (const float*)d_in[10];
  const float* out_proj_w  = (const float*)d_in[11];
  const float* pe_gauss    = (const float*)d_in[12];
  const float* point_embed = (const float*)d_in[13];
  const float* not_a_point = (const float*)d_in[14];
  const int*   labels      = (const int*)d_in[15];

  char* ws = (char*)d_ws;
  float*          zx      = (float*)(ws + WS_ZX);
  float*          xbc     = (float*)(ws + WS_XBC);
  unsigned short* packA   = (unsigned short*)(ws + WS_SCR);
  float*          Sbuf    = (float*)(ws + WS_SCR + OFF_SB);
  float*          Hbuf    = (float*)(ws + WS_SCR + OFF_HB);
  float*          Ebuf    = (float*)(ws + WS_SCR + OFF_EB);
  float*          csumBuf = (float*)(ws + WS_SCR + OFF_CS);
  unsigned short* packW   = (unsigned short*)(ws + WS_PACKW);
  float*          pemb    = (float*)(ws + WS_PEMB);
  float*          out     = (float*)d_out;

  // phase 1: LN + in_proj GEMM
  pemb_kernel<<<kB * 8, kDIM, 0, stream>>>(coords, labels, pe_gauss,
                                           point_embed, not_a_point, pemb);
  ln_pack_kernel<<<kM / 8, 256, 0, stream>>>(x, pemb, ln_w, ln_b, packA);
  {
    long n = (long)kNPAD * kDIM;
    pack_w1_kernel<<<(int)((n + 255) / 256), 256, 0, stream>>>(in_proj_w, packW);
  }
  {
    long t2 = (long)(kMT / 2) * (kNT1 / 2);
    gemm1_kernel<<<(int)((t2 + 7) / 8), 256, 0, stream>>>(packA, packW, zx);
  }
  // phase 2: conv
  {
    long n = (long)kM * kCONV;
    conv_silu_kernel<<<(int)((n + 255) / 256), 256, 0, stream>>>(zx, conv_w, conv_b, xbc);
  }
  // phase 3: chunked SSD (WMMA) — 1056 independent (b,head,chunk) blocks
  ssd_chunk_kernel<<<32 * kNC, 256, 0, stream>>>(zx, xbc, dt_bias, A_log, Dp,
                                                 Sbuf, Ebuf, csumBuf);
  ssd_state_kernel<<<32, 256, 0, stream>>>(Sbuf, Ebuf, Hbuf);
  ssd_final_kernel<<<32 * kNC, 256, 0, stream>>>(zx, xbc, Hbuf, csumBuf);
  // phase 4: RMS + out_proj GEMM (+ fused output transpose)
  rms_pack_kernel<<<kM / 8, 256, 0, stream>>>(zx, rms_w, packA);
  {
    long n = (long)kDIM * kDIN;
    pack_w2_kernel<<<(int)((n + 255) / 256), 256, 0, stream>>>(out_proj_w, packW);
  }
  {
    long t2 = (long)(kMT / 2) * (kNT2 / 2);
    gemm2_kernel<<<(int)((t2 + 7) / 8), 256, 0, stream>>>(packA, packW, out);
  }
}
